// SwinTransformerBlock_84748294685214
// MI455X (gfx1250) — matrix operand order repaired and checked by reading, not compiled, not run on hardware
//
#include <hip/hip_runtime.h>
#include <hip/hip_bf16.h>
#include <math.h>
#include <stdint.h>

// ---------------------------------------------------------------------------
// Types for gfx1250 WMMA (wave32): D(16x16 f32) = A(16x32 f16) x B(32x16 f16) + C
// ---------------------------------------------------------------------------
typedef _Float16 h8x  __attribute__((ext_vector_type(8)));
typedef _Float16 h16x __attribute__((ext_vector_type(16)));
typedef float    f8x  __attribute__((ext_vector_type(8)));

__device__ __forceinline__ f8x f8zero() {
  f8x z = {0.f,0.f,0.f,0.f,0.f,0.f,0.f,0.f};
  return z;
}
__device__ __forceinline__ h8x h8zero() {
  h8x z = {(_Float16)0,(_Float16)0,(_Float16)0,(_Float16)0,
           (_Float16)0,(_Float16)0,(_Float16)0,(_Float16)0};
  return z;
}

__device__ __forceinline__ f8x wmma_f16(h16x a, h16x b, f8x c) {
  // 8 args: (neg_a, A, neg_b, B, c_mod, C, reuse_a, reuse_b)
  return __builtin_amdgcn_wmma_f32_16x16x32_f16(false, a, false, b, (short)0, c,
                                                false, false);
}

// A-fragment (16x32 MxK f16) / B-fragment (32x16 KxN f16, staged n-major):
// two contiguous 16B LDS reads per lane compose the 16-half fragment.
__device__ __forceinline__ h16x ld_frag(const _Float16* lo, const _Float16* hi) {
  h8x l = *(const h8x*)lo;
  h8x h = *(const h8x*)hi;
  h16x r;
#pragma unroll
  for (int t = 0; t < 8; ++t) { r[t] = l[t]; r[t + 8] = h[t]; }
  return r;
}

// ---------------------------------------------------------------------------
// CDNA5 async global->LDS copy (ASYNCcnt path, no VGPR round trip).
// Generic LDS pointer low 32 bits == LDS byte offset (ISA flat aperture rule).
// ---------------------------------------------------------------------------
__device__ __forceinline__ void async_copy16(void* lds_ptr, const void* gptr) {
  uint32_t lds_off = (uint32_t)(uintptr_t)lds_ptr;
  uint64_t gaddr   = (uint64_t)(uintptr_t)gptr;
  asm volatile("global_load_async_to_lds_b128 %0, %1, off"
               :: "v"(lds_off), "v"(gaddr)
               : "memory");
}
__device__ __forceinline__ void wait_async0() {
  asm volatile("s_wait_asynccnt 0x0" ::: "memory");
}

// ---------------------------------------------------------------------------
// Geometry constants
// ---------------------------------------------------------------------------
#define HH 56
#define WW 56
#define WS 7
#define SS 3
#define NWIN 64                 // (56/7)^2 windows per image
#define NTOK 49                 // tokens per window
#define CC 256
#define NHEAD 8
#define HD 32
#define BATCH 32
#define MTOT (BATCH * NWIN * NTOK)   // 100352 rows, divisible by 128

__device__ __forceinline__ int rcls(int t) { return t < (HH - WS) ? 0 : (t < (HH - SS) ? 1 : 2); }

// ---------------------------------------------------------------------------
// f32 -> f16 cast (weights are already [N][K] row-major == B^T layout we want)
// ---------------------------------------------------------------------------
__global__ __launch_bounds__(256) void cast_f32_f16(const float* __restrict__ src,
                                                    _Float16* __restrict__ dst, int n) {
  int i = blockIdx.x * 256 + threadIdx.x;
  if (i < n) dst[i] = (_Float16)src[i];
}

// ---------------------------------------------------------------------------
// LayerNorm (wave per token). SHIFT=true additionally performs the cyclic
// shift (-3,-3) + 7x7 window partition: output row m = (b, window, intra).
// ---------------------------------------------------------------------------
template <bool SHIFT>
__global__ __launch_bounds__(256) void ln_kernel(const float* __restrict__ x,
                                                 const float* __restrict__ gw,
                                                 const float* __restrict__ gb,
                                                 _Float16* __restrict__ out) {
  int lane = threadIdx.x & 31;
  int wv   = threadIdx.x >> 5;
  int tok  = blockIdx.x * 8 + wv;          // output token index
  size_t src;
  if (SHIFT) {
    int b   = tok / (NWIN * NTOK);
    int rem = tok - b * (NWIN * NTOK);
    int w_  = rem / NTOK;
    int nn  = rem - w_ * NTOK;
    int rp  = (w_ >> 3) * WS + nn / WS;    // shifted-image coords
    int cp  = (w_ & 7) * WS + nn % WS;
    int r   = rp + SS; if (r >= HH) r -= HH;   // rolled[r'] = orig[(r'+3)%56]
    int c   = cp + SS; if (c >= WW) c -= WW;
    src = (size_t)b * (HH * WW) + r * WW + c;
  } else {
    src = (size_t)tok;
  }
  const float* xr = x + src * CC;
  float4 p0 = *(const float4*)(xr + lane * 8);
  float4 p1 = *(const float4*)(xr + lane * 8 + 4);
  float v[8] = {p0.x, p0.y, p0.z, p0.w, p1.x, p1.y, p1.z, p1.w};
  float s = 0.f, s2 = 0.f;
#pragma unroll
  for (int u = 0; u < 8; ++u) { s += v[u]; s2 += v[u] * v[u]; }
#pragma unroll
  for (int off = 16; off > 0; off >>= 1) {
    s  += __shfl_xor(s, off, 32);
    s2 += __shfl_xor(s2, off, 32);
  }
  float mean = s * (1.f / CC);
  float var  = s2 * (1.f / CC) - mean * mean;
  float rs   = rsqrtf(var + 1e-5f);
  h8x o;
#pragma unroll
  for (int u = 0; u < 8; ++u) {
    int ch = lane * 8 + u;
    o[u] = (_Float16)((v[u] - mean) * rs * gw[ch] + gb[ch]);
  }
  *(h8x*)(out + (size_t)tok * CC + lane * 8) = o;
}

// ---------------------------------------------------------------------------
// Generic WMMA GEMM: C[M,N] = A[M,K] (f16, row major) * Bt[N,K]^T (f16) + bias
// Block tile 128x64x32, 256 threads = 8 waves in a 4x2 grid, 32x32 per wave.
// Double-buffered LDS; tile k+1 streams in via GLOBAL_LOAD_ASYNC_TO_LDS_B128
// while tile k is consumed by WMMA (ASYNCcnt pipeline, one barrier/K-step).
// EPI: 0 = f16 store          (QKV)
//      1 = exact GELU -> f16  (FC1)
//      2 = window-reverse + unshift + residual(x) -> f32   (proj)
//      3 = residual(resid) -> f32                          (FC2 -> d_out)
// ---------------------------------------------------------------------------
#define BM 128
#define BN 64
#define BK 32

template <int EPI>
__global__ __launch_bounds__(256) void gemm_wmma(const _Float16* __restrict__ A,
                                                 const _Float16* __restrict__ Bt,
                                                 const float* __restrict__ bias,
                                                 _Float16* __restrict__ out16,
                                                 float* __restrict__ out32,
                                                 const float* __restrict__ resid,
                                                 int M, int N, int K) {
  __shared__ _Float16 sA[2][BM * BK];   // row major [m][k]
  __shared__ _Float16 sB[2][BN * BK];   // n-major   [n][k]  (== B^T tile)
  const int tid  = threadIdx.x;
  const int lane = tid & 31;
  const int wv   = tid >> 5;
  const int wm   = wv >> 1;          // 0..3
  const int wn   = wv & 1;           // 0..1
  const int m0   = blockIdx.x * BM;
  const int n0   = blockIdx.y * BN;
  const int half = lane >> 4;
  const int l16  = lane & 15;

  f8x acc[2][2];
  acc[0][0] = f8zero(); acc[0][1] = f8zero();
  acc[1][0] = f8zero(); acc[1][1] = f8zero();

  // per-thread staging coordinates (16B granules)
  const int ra0 = (tid)        >> 2, ca0 = (tid)        & 3;        // A granule 0
  const int ra1 = (tid + 256)  >> 2, ca1 = (tid + 256)  & 3;        // A granule 1
  const int rb  = tid >> 2,          cb  = tid & 3;                 // B granule

  auto stage = [&](int buf, int k0) {
    async_copy16(&sA[buf][ra0 * BK + ca0 * 8],
                 A + (size_t)(m0 + ra0) * K + k0 + ca0 * 8);
    async_copy16(&sA[buf][ra1 * BK + ca1 * 8],
                 A + (size_t)(m0 + ra1) * K + k0 + ca1 * 8);
    async_copy16(&sB[buf][rb * BK + cb * 8],
                 Bt + (size_t)(n0 + rb) * K + k0 + cb * 8);
  };

  stage(0, 0);
  wait_async0();
  __syncthreads();

  const int nk = K / BK;
  for (int kt = 0; kt < nk; ++kt) {
    const int buf = kt & 1;
    if (kt + 1 < nk) stage(buf ^ 1, (kt + 1) * BK);   // overlap DMA with WMMA

    h16x af[2], bf[2];
#pragma unroll
    for (int i = 0; i < 2; ++i) {
      int m = wm * 32 + i * 16 + l16;
      af[i] = ld_frag(&sA[buf][m * BK + 8 * half], &sA[buf][m * BK + 16 + 8 * half]);
    }
#pragma unroll
    for (int j = 0; j < 2; ++j) {
      int n = wn * 32 + j * 16 + l16;
      bf[j] = ld_frag(&sB[buf][n * BK + 8 * half], &sB[buf][n * BK + 16 + 8 * half]);
    }
#pragma unroll
    for (int i = 0; i < 2; ++i)
#pragma unroll
      for (int j = 0; j < 2; ++j) acc[i][j] = wmma_f16(af[i], bf[j], acc[i][j]);

    wait_async0();
    __syncthreads();
  }

  // ---- epilogue ----
#pragma unroll
  for (int i = 0; i < 2; ++i) {
#pragma unroll
    for (int j = 0; j < 2; ++j) {
      int nc = n0 + wn * 32 + j * 16 + l16;
      float bv = bias[nc];
#pragma unroll
      for (int v = 0; v < 8; ++v) {
        int mr = m0 + wm * 32 + i * 16 + v + 8 * half;
        float val = acc[i][j][v] + bv;
        if (EPI == 0) {
          out16[(size_t)mr * N + nc] = (_Float16)val;
        } else if (EPI == 1) {
          float g = 0.5f * val * (1.f + erff(val * 0.70710678118654752f));
          out16[(size_t)mr * N + nc] = (_Float16)g;
        } else if (EPI == 2) {
          int b_  = mr / (NWIN * NTOK);
          int rem = mr - b_ * (NWIN * NTOK);
          int w_  = rem / NTOK;
          int nn  = rem - w_ * NTOK;
          int rp  = (w_ >> 3) * WS + nn / WS;
          int cp  = (w_ & 7) * WS + nn % WS;
          int r = rp + SS; if (r >= HH) r -= HH;   // unshift: rolled[r'] -> orig[(r'+3)%56]
          int c = cp + SS; if (c >= WW) c -= WW;
          size_t l = (size_t)b_ * (HH * WW) + r * WW + c;
          out32[l * CC + nc] = resid[l * CC + nc] + val;
        } else {
          out32[(size_t)mr * N + nc] = resid[(size_t)mr * N + nc] + val;
        }
      }
    }
  }
}

// ---------------------------------------------------------------------------
// Windowed attention: one block per (window, head); 128 threads = 4 waves.
// S = (Q*scale) K^T + relpos-bias + shift-mask; softmax; O = P V.
// 49 padded to 64; padded keys contribute exactly zero (P cols zeroed,
// V pad rows zeroed). Q/K staged with async-to-LDS DMA; V transposed manually.
// ---------------------------------------------------------------------------
__global__ __launch_bounds__(128) void attn_kernel(const _Float16* __restrict__ qkv,
                                                   const float* __restrict__ relt,
                                                   _Float16* __restrict__ out) {
  __shared__ _Float16 sQ[64 * 32];   // [m][d]
  __shared__ _Float16 sK[64 * 32];   // [j][d]  (== (K^T)^T : n-major B tile)
  __shared__ _Float16 sV[32 * 64];   // [d][j]  (n-major B tile for PV)
  __shared__ float    sS[64 * 64];
  __shared__ _Float16 sP[64 * 64];   // [m][j] row major (A for PV)
  __shared__ float    sInv[64];

  const int tid  = threadIdx.x;
  const int lane = tid & 31;
  const int wv   = tid >> 5;
  const int half = lane >> 4;
  const int l16  = lane & 15;
  const int win  = blockIdx.x >> 3;
  const int hh   = blockIdx.x & 7;
  const size_t rowbase = (size_t)win * NTOK * (3 * CC);

  // zero-fill pad rows 49..63 of sQ/sK (15 rows x 4 granules x 2 arrays = 120)
  if (tid < 120) {
    int arr = tid / 60;
    int c   = tid - arr * 60;
    int r   = 49 + (c >> 2), ck = c & 3;
    *(h8x*)((arr ? sK : sQ) + r * 32 + ck * 8) = h8zero();
  }
  // async DMA valid Q/K rows straight into LDS (196 granules each)
#pragma unroll
  for (int it = 0; it < 2; ++it) {
    int idx = tid + it * 128;
    if (idx < NTOK * 4) {
      int r = idx >> 2, ck = idx & 3;
      const _Float16* src = qkv + rowbase + (size_t)r * (3 * CC) + hh * HD + ck * 8;
      async_copy16(sQ + r * 32 + ck * 8, src);
      async_copy16(sK + r * 32 + ck * 8, src + CC);
    }
  }
  // stage V transposed: sV[d][j], zero-pad j >= 49
#pragma unroll
  for (int it = 0; it < 2; ++it) {
    int idx = tid + it * 128;
    int j = idx >> 2, ck = idx & 3;
    h8x v = h8zero();
    if (j < NTOK)
      v = *(const h8x*)(qkv + rowbase + (size_t)j * (3 * CC) + 2 * CC + hh * HD + ck * 8);
#pragma unroll
    for (int u = 0; u < 8; ++u) sV[(ck * 8 + u) * 64 + j] = v[u];
  }
  wait_async0();
  __syncthreads();

  // S = Q K^T : wave wv owns rows [16*wv, 16*wv+16), 4 column tiles, K=32
  {
    int m = wv * 16 + l16;
    h16x a = ld_frag(sQ + m * 32 + 8 * half, sQ + m * 32 + 16 + 8 * half);
#pragma unroll
    for (int jt = 0; jt < 4; ++jt) {
      int n = jt * 16 + l16;
      h16x b = ld_frag(sK + n * 32 + 8 * half, sK + n * 32 + 16 + 8 * half);
      f8x c = f8zero();
      c = wmma_f16(a, b, c);
#pragma unroll
      for (int v = 0; v < 8; ++v)
        sS[(wv * 16 + v + 8 * half) * 64 + jt * 16 + l16] = c[v];
    }
  }
  __syncthreads();

  // softmax rows (threads 0..63); bias & shift-mask computed analytically
  if (tid < 64) {
    int i = tid;
    if (i < NTOK) {
      const float scale = 0.17677669529663687f;   // 1/sqrt(32)
      int qr = i / WS, qc = i % WS;
      int w_ = win & (NWIN - 1);
      int wr = (w_ >> 3) * WS, wc = (w_ & 7) * WS;
      int regi = 3 * rcls(wr + qr) + rcls(wc + qc);
      float sbuf[NTOK];
      float mx = -1e30f;
#pragma unroll
      for (int j = 0; j < NTOK; ++j) {
        int kr = j / WS, kc = j % WS;
        int ridx = (qr - kr + WS - 1) * (2 * WS - 1) + (qc - kc + WS - 1);
        float bias = relt[ridx * NHEAD + hh];
        int regj = 3 * rcls(wr + kr) + rcls(wc + kc);
        float s = sS[i * 64 + j] * scale + bias + (regi == regj ? 0.f : -100.f);
        sbuf[j] = s;
        mx = fmaxf(mx, s);
      }
      float sum = 0.f;
#pragma unroll
      for (int j = 0; j < NTOK; ++j) {
        float e = __expf(sbuf[j] - mx);
        sum += e;
        sP[i * 64 + j] = (_Float16)e;
      }
#pragma unroll
      for (int j = NTOK; j < 64; ++j) sP[i * 64 + j] = (_Float16)0.f;
      sInv[i] = 1.f / sum;
    } else {
#pragma unroll
      for (int j = 0; j < 64; ++j) sP[i * 64 + j] = (_Float16)0.f;
      sInv[i] = 0.f;
    }
  }
  __syncthreads();

  // O = P V : wave wv rows [16*wv, ..), 2 column tiles (head dim 32), K=64
  {
    f8x c0 = f8zero(), c1 = f8zero();
#pragma unroll
    for (int k0 = 0; k0 < 64; k0 += 32) {
      int m = wv * 16 + l16;
      h16x a  = ld_frag(sP + m * 64 + k0 + 8 * half, sP + m * 64 + k0 + 16 + 8 * half);
      h16x b0 = ld_frag(sV + (l16)      * 64 + k0 + 8 * half,
                        sV + (l16)      * 64 + k0 + 16 + 8 * half);
      h16x b1 = ld_frag(sV + (16 + l16) * 64 + k0 + 8 * half,
                        sV + (16 + l16) * 64 + k0 + 16 + 8 * half);
      c0 = wmma_f16(a, b0, c0);
      c1 = wmma_f16(a, b1, c1);
    }
#pragma unroll
    for (int v = 0; v < 8; ++v) {
      int i = wv * 16 + v + 8 * half;
      if (i < NTOK) {
        float inv = sInv[i];
        size_t o = ((size_t)win * NTOK + i) * CC + hh * HD;
        out[o + l16]      = (_Float16)(c0[v] * inv);
        out[o + 16 + l16] = (_Float16)(c1[v] * inv);
      }
    }
  }
}

// ---------------------------------------------------------------------------
// Launch
// ---------------------------------------------------------------------------
extern "C" void kernel_launch(void* const* d_in, const int* in_sizes, int n_in,
                              void* d_out, int out_size, void* d_ws, size_t ws_size,
                              hipStream_t stream) {
  const float* x      = (const float*)d_in[0];
  const float* n1w    = (const float*)d_in[1];
  const float* n1b    = (const float*)d_in[2];
  const float* qkv_w  = (const float*)d_in[3];
  const float* qkv_b  = (const float*)d_in[4];
  const float* proj_w = (const float*)d_in[5];
  const float* proj_b = (const float*)d_in[6];
  const float* relt   = (const float*)d_in[7];
  const float* n2w    = (const float*)d_in[8];
  const float* n2b    = (const float*)d_in[9];
  const float* fc1_w  = (const float*)d_in[10];
  const float* fc1_b  = (const float*)d_in[11];
  const float* fc2_w  = (const float*)d_in[12];
  const float* fc2_b  = (const float*)d_in[13];

  char* ws = (char*)d_ws;
  const size_t O_WQ  = 0;                                     // 768x256 f16
  const size_t O_WP  = O_WQ  + (size_t)768 * 256 * 2;         // 256x256 f16
  const size_t O_WF1 = O_WP  + (size_t)256 * 256 * 2;         // 1024x256 f16
  const size_t O_WF2 = O_WF1 + (size_t)1024 * 256 * 2;        // 256x1024 f16
  const size_t O_H16 = O_WF2 + (size_t)256 * 1024 * 2;        // MTOTx256 f16 (reused: attn out)
  const size_t S_H16 = (size_t)MTOT * CC * 2;
  const size_t O_QKV = O_H16 + S_H16;                         // MTOTx768 f16 (reused: y16)
  const size_t S_QKV = (size_t)MTOT * 3 * CC * 2;
  const size_t O_X2  = O_QKV + S_QKV;                         // MTOTx256 f32
  const size_t S_X2  = (size_t)MTOT * CC * 4;
  const size_t O_FC1 = O_X2 + S_X2;                           // MTOTx1024 f16

  _Float16* wq16   = (_Float16*)(ws + O_WQ);
  _Float16* wp16   = (_Float16*)(ws + O_WP);
  _Float16* wf116  = (_Float16*)(ws + O_WF1);
  _Float16* wf216  = (_Float16*)(ws + O_WF2);
  _Float16* h16buf = (_Float16*)(ws + O_H16);
  _Float16* attn16 = (_Float16*)(ws + O_H16);   // reuse (h16 dead after QKV GEMM)
  _Float16* qkv16  = (_Float16*)(ws + O_QKV);
  _Float16* y16    = (_Float16*)(ws + O_QKV);   // reuse (qkv dead after attention)
  float*    x2     = (float*)(ws + O_X2);
  _Float16* fc1buf = (_Float16*)(ws + O_FC1);
  float*    outp   = (float*)d_out;

  // weights -> f16 (layout [N][K] is already B^T)
  cast_f32_f16<<<768, 256, 0, stream>>>(qkv_w, wq16, 768 * 256);
  cast_f32_f16<<<256, 256, 0, stream>>>(proj_w, wp16, 256 * 256);
  cast_f32_f16<<<1024, 256, 0, stream>>>(fc1_w, wf116, 1024 * 256);
  cast_f32_f16<<<1024, 256, 0, stream>>>(fc2_w, wf216, 256 * 1024);

  // LN1 + shift + window partition
  ln_kernel<true><<<MTOT / 8, 256, 0, stream>>>(x, n1w, n1b, h16buf);

  // QKV GEMM: (100352 x 256) x (256 x 768)
  gemm_wmma<0><<<dim3(MTOT / BM, 768 / BN), 256, 0, stream>>>(
      h16buf, wq16, qkv_b, qkv16, nullptr, nullptr, MTOT, 768, 256);

  // windowed attention (2048 windows x 8 heads)
  attn_kernel<<<BATCH * NWIN * NHEAD, 128, 0, stream>>>(qkv16, relt, attn16);

  // proj GEMM + window-reverse + unshift + residual -> x2 (f32)
  gemm_wmma<2><<<dim3(MTOT / BM, 256 / BN), 256, 0, stream>>>(
      attn16, wp16, proj_b, nullptr, x2, x, MTOT, 256, 256);

  // LN2
  ln_kernel<false><<<MTOT / 8, 256, 0, stream>>>(x2, n2w, n2b, y16);

  // FC1 GEMM + exact GELU
  gemm_wmma<1><<<dim3(MTOT / BM, 1024 / BN), 256, 0, stream>>>(
      y16, wf116, fc1_b, fc1buf, nullptr, nullptr, MTOT, 1024, 256);

  // FC2 GEMM + residual -> d_out
  gemm_wmma<3><<<dim3(MTOT / BM, 256 / BN), 256, 0, stream>>>(
      fc1buf, wf216, fc2_b, nullptr, outp, x2, MTOT, 256, 1024);
}
